// SoftThresholdAttention_48541720380015
// MI455X (gfx1250) — compile-verified, matrix operand order checked by reading, not run on hardware
//
#include <hip/hip_runtime.h>
#include <hip/hip_bf16.h>

typedef __attribute__((ext_vector_type(16))) _Float16 v16h;
typedef __attribute__((ext_vector_type(8)))  _Float16 v8h;
typedef __attribute__((ext_vector_type(8)))  float    v8f;

#define DIM    256
#define NHEADS 4
#define HD     64
#define BATCH  2
#define SEQ    4096
#define TOKENS (BATCH * SEQ)   // 8192
#define KTILE  64              // keys staged per LDS tile
#define LDP    72              // padded LDS row length (halves): 144B stride, conflict-free

// Load a 16-lane-striped f16 fragment from a row-major matrix (global or LDS).
// Pattern (per ISA 16-bit A 16x32 / B 32x16 tables): lo = lane&15 selects the
// row (A: M, B: N via B^T); halves 0..7 = K(hi*8..hi*8+7), 8..15 = +16.
static __device__ __forceinline__ v16h ldfrag(const _Float16* row, int koff) {
  v8h a = *reinterpret_cast<const v8h*>(row + koff);
  v8h b = *reinterpret_cast<const v8h*>(row + koff + 16);
  return __builtin_shufflevector(a, b, 0,1,2,3,4,5,6,7,8,9,10,11,12,13,14,15);
}

static __device__ __forceinline__ v8f wmma_f16(v16h a, v16h b, v8f c) {
  return __builtin_amdgcn_wmma_f32_16x16x32_f16(false, a, false, b, (short)0, c, false, false);
}

// Async global -> LDS copy of one 16-byte chunk (ASYNCcnt-tracked DMA path).
static __device__ __forceinline__ void async_copy16(const _Float16* gsrc, const _Float16* ldst) {
  unsigned loff = (unsigned)(unsigned long long)ldst;  // low 32 bits = LDS offset
  asm volatile("global_load_async_to_lds_b128 %0, %1, off"
               :: "v"(loff), "v"(gsrc) : "memory");
}

static __device__ __forceinline__ void wait_async0() {
  asm volatile("s_wait_asynccnt 0x0" ::: "memory");
}

// ---------------------------------------------------------------------------
// Kernel 0: fp32 -> f16 conversion of x and the four weight matrices
// ---------------------------------------------------------------------------
__global__ __launch_bounds__(256) void st_attn_convert(
    const float* __restrict__ x,
    const float* __restrict__ qw, const float* __restrict__ kw,
    const float* __restrict__ vw, const float* __restrict__ ow,
    _Float16* __restrict__ xh, _Float16* __restrict__ wh)
{
  const int NX = TOKENS * DIM;       // 2,097,152
  const int NW = DIM * DIM;          // 65,536
  int i = blockIdx.x * blockDim.x + threadIdx.x;
  if (i < NX) {
    xh[i] = (_Float16)x[i];
  } else if (i < NX + 4 * NW) {
    int j = i - NX;
    int w = j >> 16;                 // NW == 1<<16
    int r = j & (NW - 1);
    const float* src = (w == 0) ? qw : (w == 1) ? kw : (w == 2) ? vw : ow;
    wh[j] = (_Float16)src[r];
  }
}

// ---------------------------------------------------------------------------
// Kernel 1: QKV projection  Y = x @ W^T + b   (one wave = one 16x16 tile)
//   which=0: Q (pre-scaled by DIM^-0.5), row-major f16 [token][256]
//   which=1: K, row-major f16 [token][256]
//   which=2: V, stored transposed per head: Vth[b][h][d][token] f16
// ---------------------------------------------------------------------------
__global__ __launch_bounds__(128) void st_attn_qkv(
    const _Float16* __restrict__ xh, const _Float16* __restrict__ wh,
    const float* __restrict__ qbias, const float* __restrict__ kbias,
    const float* __restrict__ vbias,
    _Float16* __restrict__ Qh, _Float16* __restrict__ Kh, _Float16* __restrict__ Vth)
{
  const int lane = threadIdx.x & 31;
  const int lo = lane & 15, hi = lane >> 4;
  const int which = blockIdx.y;
  const _Float16* w = wh + which * (DIM * DIM);
  const float* bias = (which == 0) ? qbias : (which == 1) ? kbias : vbias;

  const int tile = blockIdx.x * (blockDim.x >> 5) + (threadIdx.x >> 5); // < 8192
  const int m0 = (tile >> 4) * 16;   // token tile
  const int n0 = (tile & 15) * 16;   // out-channel tile

  const _Float16* arow = xh + (m0 + lo) * DIM;  // A: x rows (M=token, K=chan)
  const _Float16* brow = w  + (n0 + lo) * DIM;  // B^T: W rows (N=outchan, K=chan)

  v8f acc = {0.f,0.f,0.f,0.f,0.f,0.f,0.f,0.f};
  #pragma unroll
  for (int k = 0; k < DIM; k += 32)
    acc = wmma_f16(ldfrag(arow, k + hi * 8), ldfrag(brow, k + hi * 8), acc);

  const float bv  = bias[n0 + lo];
  const float osc = (which == 0) ? 0.0625f : 1.0f;  // fold attn scale into Q

  if (which < 2) {
    _Float16* out = (which == 0) ? Qh : Kh;
    #pragma unroll
    for (int v = 0; v < 8; ++v) {
      int tok = m0 + v + 8 * hi;                    // C layout: M = v + 8*hi
      out[tok * DIM + n0 + lo] = (_Float16)((acc[v] + bv) * osc);
    }
  } else {
    // C-tile per lane holds 8 consecutive tokens at one fixed channel ->
    // a single contiguous 16B store into the transposed V layout.
    int chan = n0 + lo;
    int hh = chan >> 6, d = chan & (HD - 1);
    int bb = m0 >> 12;
    int tok0 = (m0 & (SEQ - 1)) + 8 * hi;
    v8h p;
    #pragma unroll
    for (int v = 0; v < 8; ++v) p[v] = (_Float16)(acc[v] + bv);
    *reinterpret_cast<v8h*>(Vth + ((bb * NHEADS + hh) * HD + d) * SEQ + tok0) = p;
  }
}

// ---------------------------------------------------------------------------
// Kernel 2: fused soft-threshold flash attention (transposed formulation).
// 8 waves/block share one (b,h); K/V tiles are staged cooperatively into LDS
// with double-buffered GLOBAL_LOAD_ASYNC_TO_LDS_B128 (ASYNCcnt), so each K/V
// tile is fetched once per block instead of once per wave (8x VMEM cut).
// S^T = K*Q^T keeps the softmax reduction per-lane, and the stacked S^T
// C-tiles ARE the B-fragment of P^T for the O^T += V^T*P^T WMMA.
// ---------------------------------------------------------------------------
__global__ __launch_bounds__(256) void st_attn_fa(
    const _Float16* __restrict__ Qh, const _Float16* __restrict__ Kh,
    const _Float16* __restrict__ Vth, const float* __restrict__ taup,
    _Float16* __restrict__ AOh)
{
  __shared__ _Float16 sk[2][KTILE][LDP];   // K tile: [key][dh], 18 KB
  __shared__ _Float16 sv[2][KTILE][LDP];   // V tile: [dh][key], 18 KB

  const int tid  = threadIdx.x;
  const int lane = tid & 31;
  const int lo = lane & 15, hi = lane >> 4;
  const int wid = blockIdx.x * (blockDim.x >> 5) + (tid >> 5); // < 2048
  const int bb = wid >> 10;              // / (NHEADS * SEQ/16)
  const int hh = (wid >> 8) & (NHEADS - 1);
  const int q0 = (wid & 255) * 16;
  const float tau = taup[0];
  const float LOG2E = 1.44269504088896f;

  // Q B-fragments (B^T = Q rows), resident for the whole key loop
  const _Float16* qrow = Qh + (bb * SEQ + q0 + lo) * DIM + hh * HD;
  v16h qf0 = ldfrag(qrow,      hi * 8);   // dh 0..31
  v16h qf1 = ldfrag(qrow + 32, hi * 8);   // dh 32..63

  // copy-thread mapping: each thread moves rows {r, r+32}, 16B chunk c
  const int cr = tid >> 3;               // 0..31
  const int cc = (tid & 7) * 8;          // half offset 0..56
  const _Float16* kcopy = Kh + (bb * SEQ) * DIM + hh * HD;        // + key*DIM
  const _Float16* vcopy = Vth + (bb * NHEADS + hh) * HD * SEQ;    // + d*SEQ

  const v8f z = {0.f,0.f,0.f,0.f,0.f,0.f,0.f,0.f};
  v8f o0 = z, o1 = z, o2 = z, o3 = z;    // O^T tiles, d-chunks {0,16,32,48}
  float m = 0.f, lsum = 0.f;             // gated scores are >= 0, so m0 = 0

  // prologue: stage tile 0 into buffer 0
  {
    async_copy16(kcopy + (0 + cr) * DIM + cc,       &sk[0][cr][cc]);
    async_copy16(kcopy + (0 + cr + 32) * DIM + cc,  &sk[0][cr + 32][cc]);
    async_copy16(vcopy + cr * SEQ + 0 + cc,         &sv[0][cr][cc]);
    async_copy16(vcopy + (cr + 32) * SEQ + 0 + cc,  &sv[0][cr + 32][cc]);
  }

  for (int kb = 0, it = 0; kb < SEQ; kb += KTILE, ++it) {
    const int buf = it & 1;
    wait_async0();        // this wave's async writes done
    __syncthreads();      // => everyone's writes done; prev reads of buf^1 done
    if (kb + KTILE < SEQ) {
      const int nb = buf ^ 1, nk = kb + KTILE;
      async_copy16(kcopy + (nk + cr) * DIM + cc,      &sk[nb][cr][cc]);
      async_copy16(kcopy + (nk + cr + 32) * DIM + cc, &sk[nb][cr + 32][cc]);
      async_copy16(vcopy + cr * SEQ + nk + cc,        &sv[nb][cr][cc]);
      async_copy16(vcopy + (cr + 32) * SEQ + nk + cc, &sv[nb][cr + 32][cc]);
    }

    #pragma unroll
    for (int kl = 0; kl < KTILE; kl += 32) {
      // S^T tiles: keys kl..kl+15 (st0) and kl+16..kl+31 (st1) from LDS
      const _Float16* kr0 = &sk[buf][kl + lo][0];
      const _Float16* kr1 = &sk[buf][kl + 16 + lo][0];
      v8f st0 = wmma_f16(ldfrag(kr0,      hi * 8), qf0, z);
      st0     = wmma_f16(ldfrag(kr0 + 32, hi * 8), qf1, st0);
      v8f st1 = wmma_f16(ldfrag(kr1,      hi * 8), qf0, z);
      st1     = wmma_f16(ldfrag(kr1 + 32, hi * 8), qf1, st1);

      // gate: relu(where(|s|>tau, s, 0)) ; per-lane block max
      float g[16];
      float tmax = 0.f;
      #pragma unroll
      for (int i = 0; i < 8; ++i) {
        float s = st0[i];
        s = (fabsf(s) > tau) ? s : 0.f;
        s = fmaxf(s, 0.f);
        g[i] = s; tmax = fmaxf(tmax, s);
        s = st1[i];
        s = (fabsf(s) > tau) ? s : 0.f;
        s = fmaxf(s, 0.f);
        g[8 + i] = s; tmax = fmaxf(tmax, s);
      }
      // lanes n and n+16 hold complementary key halves of the same query
      tmax = fmaxf(tmax, __shfl_xor(tmax, 16, 32));
      float mnew  = fmaxf(m, tmax);
      float alpha = exp2f((m - mnew) * LOG2E);

      // exp + pack straight into the 32x16 P^T B-fragment (layout identity)
      v16h pb;
      float psum = 0.f;
      #pragma unroll
      for (int i = 0; i < 16; ++i) {
        float p = exp2f((g[i] - mnew) * LOG2E);
        psum += p;
        pb[i] = (_Float16)p;
      }
      lsum = lsum * alpha + psum;
      o0 = o0 * alpha; o1 = o1 * alpha; o2 = o2 * alpha; o3 = o3 * alpha;

      // O^T += V^T-chunk (16d x 32keys) @ P^T (32keys x 16q), V from LDS
      o0 = wmma_f16(ldfrag(&sv[buf][ 0 + lo][kl], hi * 8), pb, o0);
      o1 = wmma_f16(ldfrag(&sv[buf][16 + lo][kl], hi * 8), pb, o1);
      o2 = wmma_f16(ldfrag(&sv[buf][32 + lo][kl], hi * 8), pb, o2);
      o3 = wmma_f16(ldfrag(&sv[buf][48 + lo][kl], hi * 8), pb, o3);
      m = mnew;
    }
  }

  lsum += __shfl_xor(lsum, 16, 32);       // combine the two key-half sums
  float inv = 1.0f / lsum;

  // O^T lane holds 8 consecutive d-channels of one token -> 16B f16 stores
  _Float16* ao = AOh + (bb * SEQ + q0 + lo) * DIM + hh * HD;
  v8f ot[4] = {o0, o1, o2, o3};
  #pragma unroll
  for (int t = 0; t < 4; ++t) {
    v8h r;
    #pragma unroll
    for (int v = 0; v < 8; ++v) r[v] = (_Float16)(ot[t][v] * inv);
    *reinterpret_cast<v8h*>(ao + t * 16 + hi * 8) = r;
  }
}

// ---------------------------------------------------------------------------
// Kernel 3: output projection  out = AO @ o_w^T + o_b  (f32 output)
// ---------------------------------------------------------------------------
__global__ __launch_bounds__(128) void st_attn_oproj(
    const _Float16* __restrict__ aoh, const _Float16* __restrict__ woh,
    const float* __restrict__ obias, float* __restrict__ out)
{
  const int lane = threadIdx.x & 31;
  const int lo = lane & 15, hi = lane >> 4;
  const int tile = blockIdx.x * (blockDim.x >> 5) + (threadIdx.x >> 5);
  const int m0 = (tile >> 4) * 16;
  const int n0 = (tile & 15) * 16;

  const _Float16* arow = aoh + (m0 + lo) * DIM;
  const _Float16* brow = woh + (n0 + lo) * DIM;

  v8f acc = {0.f,0.f,0.f,0.f,0.f,0.f,0.f,0.f};
  #pragma unroll
  for (int k = 0; k < DIM; k += 32)
    acc = wmma_f16(ldfrag(arow, k + hi * 8), ldfrag(brow, k + hi * 8), acc);

  const float bv = obias[n0 + lo];
  #pragma unroll
  for (int v = 0; v < 8; ++v)
    out[(m0 + v + 8 * hi) * DIM + n0 + lo] = acc[v] + bv;
}

// ---------------------------------------------------------------------------
extern "C" void kernel_launch(void* const* d_in, const int* in_sizes, int n_in,
                              void* d_out, int out_size, void* d_ws, size_t ws_size,
                              hipStream_t stream)
{
  (void)in_sizes; (void)n_in; (void)out_size; (void)ws_size;
  const float* x   = (const float*)d_in[0];
  const float* qw  = (const float*)d_in[1];
  const float* qb  = (const float*)d_in[2];
  const float* kw  = (const float*)d_in[3];
  const float* kb  = (const float*)d_in[4];
  const float* vw  = (const float*)d_in[5];
  const float* vb  = (const float*)d_in[6];
  const float* ow  = (const float*)d_in[7];
  const float* ob  = (const float*)d_in[8];
  const float* tau = (const float*)d_in[9];
  float* out = (float*)d_out;

  const size_t NX = (size_t)TOKENS * DIM;  // 2,097,152 halves
  const size_t NW = (size_t)DIM * DIM;     //    65,536 halves
  _Float16* ws  = (_Float16*)d_ws;
  _Float16* xh  = ws;             // x in f16
  _Float16* wh  = xh + NX;        // q,k,v,o weights f16 (4*NW)
  _Float16* Qh  = wh + 4 * NW;    // Q (pre-scaled) row-major f16
  _Float16* Kh  = Qh + NX;        // K row-major f16
  _Float16* Vth = Kh + NX;        // V transposed per head [b][h][d][n] f16
  _Float16* AOh = Vth + NX;       // attention output f16 [token][256]
  // total ~21.5 MB of workspace

  {
    int total = (int)(NX + 4 * NW);
    st_attn_convert<<<(total + 255) / 256, 256, 0, stream>>>(x, qw, kw, vw, ow, xh, wh);
  }
  // 8192 tiles per matrix, 4 waves/block -> 2048 blocks; y selects Q/K/V
  st_attn_qkv<<<dim3(2048, 3), 128, 0, stream>>>(xh, wh, qb, kb, vb, Qh, Kh, Vth);
  // 2048 waves (B*H*N/16), 8 waves/block sharing one (b,h) via LDS staging
  st_attn_fa<<<256, 256, 0, stream>>>(Qh, Kh, Vth, tau, AOh);
  // output projection
  st_attn_oproj<<<2048, 128, 0, stream>>>(AOh, wh + 3 * NW, ob, out);
}
